// IrregularityAwareAttention_19481971654936
// MI455X (gfx1250) — matrix operand order repaired and checked by reading, not compiled, not run on hardware
//
#include <hip/hip_runtime.h>
#include <hip/hip_bf16.h>
#include <stdint.h>

#define B_  8
#define N_  2048
#define D_  512
#define DA_ 64

typedef __bf16 bf16;
typedef __attribute__((ext_vector_type(16))) __bf16 bf16x16;
typedef __attribute__((ext_vector_type(8)))  __bf16 bf16x8;
typedef __attribute__((ext_vector_type(4)))  __bf16 bf16x4;
typedef __attribute__((ext_vector_type(8)))  float  f32x8;
typedef __attribute__((ext_vector_type(4)))  float  f32x4;
typedef int gcc_i32x4 __attribute__((vector_size(16)));   // matches V4i builtin param

#if defined(__gfx1250__) &&                                                   \
    __has_builtin(__builtin_amdgcn_global_load_async_to_lds_b128) &&          \
    __has_builtin(__builtin_amdgcn_s_wait_asynccnt)
#define HAVE_ASYNC_COPY 1
#else
#define HAVE_ASYNC_COPY 0
#endif

__device__ __forceinline__ f32x8 zero8() {
    f32x8 z;
#pragma unroll
    for (int i = 0; i < 8; i++) z[i] = 0.0f;
    return z;
}

// D = A(16x32 bf16) * B(32x16 bf16) + C(16x16 f32)
__device__ __forceinline__ f32x8 wmma_bf16(bf16x16 a, bf16x16 b, f32x8 c) {
    return __builtin_amdgcn_wmma_f32_16x16x32_bf16(
        /*neg_a=*/false, a, /*neg_b=*/false, b,
        /*c_mod=*/(short)0, c, /*reuse_a=*/false, /*reuse_b=*/false);
}

// A-fragment (16x32, 16-bit) from row-major bf16 storage (LDS or global).
// Lane: m = lane&15, half h = lane>>4.  Elements 0..15 = K[h*16 .. h*16+15]
// (contiguous order, matching the B fragments).
__device__ __forceinline__ bf16x16 load_a_frag(const bf16* base, int ld,
                                               int m, int kbase, int h) {
    const bf16* p0 = base + (size_t)m * ld + kbase + h * 16;
    const bf16* p1 = p0 + 8;
    bf16x8 lo = *(const bf16x8*)p0;
    bf16x8 hi = *(const bf16x8*)p1;
    bf16x16 r;
#pragma unroll
    for (int i = 0; i < 8; i++) { r[i] = lo[i]; r[8 + i] = hi[i]; }
    return r;
}

// A-fragment built from fp32 source (converted to bf16 on the fly).
__device__ __forceinline__ bf16x16 load_a_frag_f32(const float* rowp,
                                                   int kbase, int h) {
    const float* p0 = rowp + kbase + h * 16;
    const float* p1 = p0 + 8;
    bf16x16 r;
#pragma unroll
    for (int i = 0; i < 8; i++) { r[i] = (bf16)p0[i]; r[8 + i] = (bf16)p1[i]; }
    return r;
}

// B-fragment (32x16, 16-bit): lane n = lane&15, half h = lane>>4.
// VGPR v holds K = h*16 + 2v, 2v+1 -> one contiguous 16-element (32B) chunk.
__device__ __forceinline__ bf16x16 load_b_frag_f32(const float* p) {
    bf16x16 r;
#pragma unroll
    for (int i = 0; i < 16; i++) r[i] = (bf16)p[i];
    return r;
}

// ---------------------------------------------------------------------------
// Kernel 1: convert + transpose weights:  wt[w][o][d] = (bf16) W_w[d][o]
// ---------------------------------------------------------------------------
__global__ __launch_bounds__(256) void convert_weights(
    const float* __restrict__ Wq, const float* __restrict__ Wk,
    const float* __restrict__ Wv, bf16* __restrict__ wt) {
    int idx = blockIdx.x * 256 + threadIdx.x;       // over 3*512*512
    if (idx >= 3 * D_ * D_) return;
    int w = idx / (D_ * D_);
    int r = idx % (D_ * D_);
    int o = r / D_;
    int d = r % D_;
    const float* W = (w == 0) ? Wq : ((w == 1) ? Wk : Wv);
    wt[idx] = (bf16)W[(size_t)d * D_ + o];
}

// ---------------------------------------------------------------------------
// Kernel 2: fused QKV projection GEMM (bf16 WMMA, fp32 accumulate)
//   q,k row-major bf16 [B*N, D];  v stored transposed bf16 [B, D, N]
//   grid = (16384/64, 512/32), block = 256 (8 waves). Each wave owns one
//   16x16 output tile per weight matrix; x tile staged 64x256 in LDS.
//   q/k output tiles bounced through LDS for coalesced b128 stores.
// ---------------------------------------------------------------------------
__global__ __launch_bounds__(256) void qkv_proj(
    const float* __restrict__ x, const bf16* __restrict__ wt,
    const float* __restrict__ bq, const float* __restrict__ bk,
    const float* __restrict__ bv,
    bf16* __restrict__ qbuf, bf16* __restrict__ kbuf,
    bf16* __restrict__ vtbuf) {
    __shared__ bf16 XT[64 * 256];                   // 32 KB
    __shared__ bf16 QO[64 * 32];                    // 4 KB
    __shared__ bf16 KO[64 * 32];                    // 4 KB

    const int tid  = threadIdx.x;
    const int wave = tid >> 5;
    const int lane = tid & 31;
    const int ln   = lane & 15;
    const int h    = lane >> 4;
    const int row0 = blockIdx.x * 64;
    const int col0 = blockIdx.y * 32;
    const int mSub = wave >> 1;
    const int nSub = wave & 1;

    const int m = mSub * 16 + ln;                   // local row in tile
    const int o = col0 + nSub * 16 + ln;            // output column

    const bf16* wq = wt + (size_t)0 * D_ * D_ + (size_t)o * D_;
    const bf16* wk = wt + (size_t)1 * D_ * D_ + (size_t)o * D_;
    const bf16* wv = wt + (size_t)2 * D_ * D_ + (size_t)o * D_;

    f32x8 accQ = zero8(), accK = zero8(), accV = zero8();

    for (int kh = 0; kh < 2; kh++) {                // two 256-wide K panels
        __syncthreads();
        // stage x[row0:row0+64, kh*256 : kh*256+256] as bf16
        const float* src = x + (size_t)row0 * D_ + kh * 256;
        for (int it = tid; it < 64 * 256 / 4; it += 256) {
            int e    = it * 4;
            int r    = e >> 8;
            int dloc = e & 255;
            f32x4 v = *(const f32x4*)(src + (size_t)r * D_ + dloc);
            bf16x4 o4;
#pragma unroll
            for (int j = 0; j < 4; j++) o4[j] = (bf16)v[j];
            *(bf16x4*)&XT[e] = o4;
        }
        __syncthreads();

#pragma unroll
        for (int c = 0; c < 8; c++) {
            bf16x16 a = load_a_frag(XT, 256, m, c * 32, h);
            int koff = kh * 256 + c * 32 + h * 16;
            bf16x16 b0 = *(const bf16x16*)(wq + koff);
            accQ = wmma_bf16(a, b0, accQ);
            bf16x16 b1 = *(const bf16x16*)(wk + koff);
            accK = wmma_bf16(a, b1, accK);
            bf16x16 b2 = *(const bf16x16*)(wv + koff);
            accV = wmma_bf16(a, b2, accV);
        }
    }

    const float biasq = bq[o], biask = bk[o], biasv = bv[o];

    // q, k: repack C-fragments (lane->col, VGPR->row) through LDS so global
    // stores are coalesced 16B instead of 16 scattered 2B stores per lane.
#pragma unroll
    for (int i = 0; i < 8; i++) {
        int lr = mSub * 16 + h * 8 + i;
        QO[lr * 32 + nSub * 16 + ln] = (bf16)(accQ[i] + biasq);
        KO[lr * 32 + nSub * 16 + ln] = (bf16)(accK[i] + biask);
    }
    // v transposed: 8 consecutive rows for fixed o -> one 16B packed store
    {
        int grow0 = row0 + mSub * 16 + h * 8;
        int batch = grow0 / N_;
        int np    = grow0 % N_;
        bf16x8 pv;
#pragma unroll
        for (int i = 0; i < 8; i++) pv[i] = (bf16)(accV[i] + biasv);
        *(bf16x8*)(vtbuf + ((size_t)batch * D_ + o) * N_ + np) = pv;
    }
    __syncthreads();
    {
        int e  = tid * 8;                           // 64*32 / 256 = 8 each
        int r  = e >> 5;
        int cc = e & 31;
        bf16x8 vq = *(const bf16x8*)&QO[e];
        bf16x8 vk = *(const bf16x8*)&KO[e];
        *(bf16x8*)(qbuf + (size_t)(row0 + r) * D_ + col0 + cc) = vq;
        *(bf16x8*)(kbuf + (size_t)(row0 + r) * D_ + col0 + cc) = vk;
    }
}

// ---------------------------------------------------------------------------
// Kernel 3: flash attention.
//   grid = (N/16, B), block = 128 (4 waves). Block owns 16 query rows.
//   Per 64-key block: wave w computes S tile for keys [kb*64+16w, +16),
//   online softmax across waves via LDS, P staged bf16 in LDS, then wave w
//   accumulates O columns [128w, 128w+128) with P@Vt WMMAs.
//   Q tile staged via async global->LDS copy when available (ASYNCcnt path).
// ---------------------------------------------------------------------------
__global__ __launch_bounds__(128) void attn_kernel(
    const bf16* __restrict__ qbuf, const bf16* __restrict__ kbuf,
    const bf16* __restrict__ vtbuf,
    const float* __restrict__ qaux, const float* __restrict__ kaux,
    const int* __restrict__ adj,
    const float* __restrict__ cmerge, const float* __restrict__ thrp,
    float* __restrict__ out) {
    __shared__ bf16  Qlds[16 * D_];                 // 16 KB
    __shared__ bf16  Plds[16 * 64];                 // 2 KB
    __shared__ float wmaxs[4][16];
    __shared__ float wsums[4][16];
    __shared__ float rowM[16];
    __shared__ float rowL[16];

    const int tid   = threadIdx.x;
    const int wave  = tid >> 5;
    const int lane  = tid & 31;
    const int ln    = lane & 15;
    const int h     = lane >> 4;
    const int qt    = blockIdx.x;                   // 0..127
    const int b     = blockIdx.y;                   // 0..7
    const int qrow0 = qt * 16;
    const size_t rowbase = (size_t)b * N_ + qrow0;

    // stage Q tile (16 x 512 bf16) into LDS
    {
        const bf16* src = qbuf + rowbase * D_;
#if HAVE_ASYNC_COPY
        for (int it = tid; it < 16 * D_ / 8; it += 128) {
            __builtin_amdgcn_global_load_async_to_lds_b128(
                (gcc_i32x4*)(uintptr_t)(src + (size_t)it * 8),
                (gcc_i32x4*)(uintptr_t)(Qlds + (size_t)it * 8),
                0, 0);
        }
        __builtin_amdgcn_s_wait_asynccnt(0);
#else
        for (int it = tid; it < 16 * D_ / 8; it += 128)
            ((bf16x8*)Qlds)[it] = ((const bf16x8*)src)[it];
#endif
    }
    // query_aux A-fragments (DA=64 -> two K=32 chunks), kept in VGPRs
    bf16x16 qaF[2];
    {
        const float* rowp = qaux + (rowbase + ln) * DA_;
#pragma unroll
        for (int c = 0; c < 2; c++) qaF[c] = load_a_frag_f32(rowp, c * 32, h);
    }
    if (tid < 16) { rowM[tid] = -__builtin_inff(); rowL[tid] = 0.0f; }
    __syncthreads();

    const float cM     = cmerge[0];
    const float thr    = thrp[0];
    const float sscale = 0.044194173824159216f;     // 1/sqrt(512)
    const float ascale = 0.125f;                    // 1/sqrt(64)
    const float MASKV  = -3.402823466e+38f;         // finfo(f32).min

    f32x8 O[8];
#pragma unroll
    for (int t = 0; t < 8; t++) O[t] = zero8();

    for (int kb = 0; kb < N_ / 64; kb++) {
        const int key0 = kb * 64 + wave * 16;
        const int keyn = key0 + ln;

        // prefetch next adjacency tile into cache hierarchy
        if (kb + 1 < N_ / 64)
            __builtin_prefetch(
                &adj[((size_t)b * N_ + qrow0 + h * 8) * N_ + (kb + 1) * 64 +
                     wave * 16 + ln], 0, 1);

        // ---- S = Q K^T (16 WMMAs over D=512) ----
        f32x8 S = zero8();
        const bf16* krow = kbuf + ((size_t)b * N_ + keyn) * D_;
#pragma unroll
        for (int c = 0; c < 16; c++) {
            bf16x16 a  = load_a_frag(Qlds, D_, ln, c * 32, h);
            bf16x16 bb = *(const bf16x16*)(krow + c * 32 + h * 16);
            S = wmma_bf16(a, bb, S);
        }
        // ---- aux = Qa Ka^T (2 WMMAs over DA=64) ----
        f32x8 A = zero8();
        const float* karow = kaux + ((size_t)b * N_ + keyn) * DA_;
#pragma unroll
        for (int c = 0; c < 2; c++) {
            bf16x16 bb = load_b_frag_f32(karow + c * 32 + h * 16);
            A = wmma_bf16(qaF[c], bb, A);
        }

        // ---- blend + adjacency mask (elementwise, fp32) ----
        float s8[8];
#pragma unroll
        for (int i = 0; i < 8; i++) {
            float s  = S[i] * sscale;
            float av = A[i] * ascale;
            if (s > thr && av != 0.0f) s = (1.0f - cM) * s + cM * av;
            int mq = qrow0 + h * 8 + i;
            int am = adj[((size_t)b * N_ + mq) * N_ + keyn];
            if (am == 0) s = MASKV;
            s8[i] = s;
        }

        // ---- per-wave row max over 16 key columns ----
#pragma unroll
        for (int i = 0; i < 8; i++) {
            float v = s8[i];
            for (int d = 1; d < 16; d <<= 1)
                v = fmaxf(v, __shfl_xor(v, d, 16));
            if (ln == 0) wmaxs[wave][h * 8 + i] = v;
        }
        __syncthreads();

        // ---- new row max, rescale O, exponentiate, partial row sum ----
#pragma unroll
        for (int i = 0; i < 8; i++) {
            int r = h * 8 + i;
            float om = rowM[r];
            float nm = fmaxf(om,
                fmaxf(fmaxf(wmaxs[0][r], wmaxs[1][r]),
                      fmaxf(wmaxs[2][r], wmaxs[3][r])));
            float p = __expf(s8[i] - nm);
            float alpha = __expf(om - nm);
#pragma unroll
            for (int t = 0; t < 8; t++) O[t][i] *= alpha;
            Plds[r * 64 + wave * 16 + ln] = (bf16)p;
            float v = p;
            for (int d = 1; d < 16; d <<= 1) v += __shfl_xor(v, d, 16);
            if (ln == 0) wsums[wave][r] = v;
        }
        __syncthreads();

        // one wave folds the block into the running stats
        if (wave == 0 && lane < 16) {
            int r = lane;
            float om = rowM[r];
            float nm = fmaxf(om,
                fmaxf(fmaxf(wmaxs[0][r], wmaxs[1][r]),
                      fmaxf(wmaxs[2][r], wmaxs[3][r])));
            float alpha = __expf(om - nm);
            rowL[r] = rowL[r] * alpha +
                      wsums[0][r] + wsums[1][r] + wsums[2][r] + wsums[3][r];
            rowM[r] = nm;
        }

        // ---- O += P @ V  (16 WMMAs: 8 n-tiles x 2 K-chunks) ----
#pragma unroll
        for (int c2 = 0; c2 < 2; c2++) {
            bf16x16 a = load_a_frag(Plds, 64, ln, c2 * 32, h);
#pragma unroll
            for (int t = 0; t < 8; t++) {
                int n = wave * 128 + t * 16 + ln;
                const bf16* vrow = vtbuf + ((size_t)b * D_ + n) * N_ +
                                   kb * 64 + c2 * 32 + h * 16;
                bf16x16 bb = *(const bf16x16*)vrow;
                O[t] = wmma_bf16(a, bb, O[t]);
            }
        }
        __syncthreads();
    }

    // ---- normalize and store fp32 output ----
#pragma unroll
    for (int i = 0; i < 8; i++) {
        int r = h * 8 + i;
        float inv = 1.0f / rowL[r];
        int mq = qrow0 + r;
#pragma unroll
        for (int t = 0; t < 8; t++) {
            int d = wave * 128 + t * 16 + ln;
            out[((size_t)b * N_ + mq) * D_ + d] = O[t][i] * inv;
        }
    }
}

// ---------------------------------------------------------------------------
extern "C" void kernel_launch(void* const* d_in, const int* in_sizes, int n_in,
                              void* d_out, int out_size, void* d_ws,
                              size_t ws_size, hipStream_t stream) {
    const float* x    = (const float*)d_in[0];
    const float* qa   = (const float*)d_in[1];
    const float* ka   = (const float*)d_in[2];
    const int*   adj  = (const int*)d_in[3];
    const float* cm   = (const float*)d_in[4];
    const float* Wq   = (const float*)d_in[5];
    const float* bq   = (const float*)d_in[6];
    const float* Wk   = (const float*)d_in[7];
    const float* bk   = (const float*)d_in[8];
    const float* Wv   = (const float*)d_in[9];
    const float* bv   = (const float*)d_in[10];
    const float* thr  = (const float*)d_in[11];
    float* out = (float*)d_out;

    char* ws = (char*)d_ws;
    bf16* qbuf  = (bf16*)(ws + (size_t)0);
    bf16* kbuf  = (bf16*)(ws + (size_t)16 * 1024 * 1024);
    bf16* vtbuf = (bf16*)(ws + (size_t)32 * 1024 * 1024);
    bf16* wt    = (bf16*)(ws + (size_t)48 * 1024 * 1024);

    convert_weights<<<(3 * D_ * D_ + 255) / 256, 256, 0, stream>>>(Wq, Wk, Wv,
                                                                   wt);
    qkv_proj<<<dim3((B_ * N_) / 64, D_ / 32), 256, 0, stream>>>(
        x, wt, bq, bk, bv, qbuf, kbuf, vtbuf);
    attn_kernel<<<dim3(N_ / 16, B_), 128, 0, stream>>>(
        qbuf, kbuf, vtbuf, qa, ka, adj, cm, thr, out);
}
